// NMSLayer_40261023433334
// MI455X (gfx1250) — 8-bit matrix operand order repaired and checked by reading, compiled, not run on hardware
//
#include <hip/hip_runtime.h>
#include <stdint.h>

#define BATCH 8
#define N     4096
#define W     128
#define CONF_TH 0.65f
#define NMS_TH  0.55f

typedef int v8i __attribute__((ext_vector_type(8)));

__device__ __forceinline__ unsigned sortable(unsigned u) {
  return u ^ ((unsigned)((int)u >> 31) | 0x80000000u);
}

__global__ void prep_kernel(const float* __restrict__ x,
                            float4* __restrict__ boxes_u, int* __restrict__ cls_u,
                            float* __restrict__ obj_u, float* __restrict__ conf_u,
                            unsigned long long* __restrict__ keys) {
  int wave = threadIdx.x >> 5;
  int lane = threadIdx.x & 31;
  int g = blockIdx.x * 4 + wave;
  if (g >= BATCH * N) return;
  const float* xp = x + (size_t)g * 85;
  float v0 = xp[lane];
  float v1 = xp[32 + lane];
  float v2 = (lane < 21) ? xp[64 + lane] : 0.0f;

  unsigned long long best = 0;
  if (lane >= 5) {
    unsigned long long p = ((unsigned long long)sortable(__float_as_uint(v0)) << 7)
                         | (unsigned)(127 - (lane - 5));
    best = p;
  }
  {
    unsigned long long p = ((unsigned long long)sortable(__float_as_uint(v1)) << 7)
                         | (unsigned)(127 - (lane + 27));
    best = p > best ? p : best;
  }
  if (lane < 21) {
    unsigned long long p = ((unsigned long long)sortable(__float_as_uint(v2)) << 7)
                         | (unsigned)(127 - (lane + 59));
    best = p > best ? p : best;
  }
#pragma unroll
  for (int off = 16; off > 0; off >>= 1) {
    unsigned long long o = __shfl_xor(best, off, 32);
    best = o > best ? o : best;
  }
  float cx = __shfl(v0, 0, 32);
  float cy = __shfl(v0, 1, 32);
  float wd = __shfl(v0, 2, 32);
  float ht = __shfl(v0, 3, 32);
  float ob = __shfl(v0, 4, 32);
  if (lane == 0) {
    unsigned s = (unsigned)(best >> 7);
    int cls = 127 - (int)(best & 0x7F);
    unsigned u = (s & 0x80000000u) ? (s ^ 0x80000000u) : ~s;
    float conf = __uint_as_float(u);
    bool valid = ob > CONF_TH;
    float4 bx;
    bx.x = cx - wd * 0.5f; bx.y = cy - ht * 0.5f;
    bx.z = cx + wd * 0.5f; bx.w = cy + ht * 0.5f;
    boxes_u[g] = bx;
    cls_u[g]   = cls;
    obj_u[g]   = ob;
    conf_u[g]  = conf;
    float skey = valid ? conf : -__builtin_inff();
    unsigned ks = sortable(__float_as_uint(skey));
    keys[g] = ((unsigned long long)(~ks) << 32) | (unsigned)(g & (N - 1));
  }
}

__global__ void __launch_bounds__(1024) sort_kernel(
    const unsigned long long* __restrict__ keysIn,
    const float4* __restrict__ boxes_u, const int* __restrict__ cls_u,
    const float* __restrict__ obj_u, const float* __restrict__ conf_u,
    float4* __restrict__ boxes_s, int* __restrict__ cls_s,
    float* __restrict__ obj_s, float* __restrict__ conf_s,
    int* __restrict__ Vout) {
  __shared__ unsigned long long k[N];
  __shared__ int cnt;
  int b = blockIdx.x;
  int tid = threadIdx.x;
  for (int i = tid; i < N; i += 1024) k[i] = keysIn[b * N + i];
  if (tid == 0) cnt = 0;
  __syncthreads();
  for (int kk = 2; kk <= N; kk <<= 1) {
    for (int j = kk >> 1; j > 0; j >>= 1) {
      for (int i = tid; i < N; i += 1024) {
        int ixj = i ^ j;
        if (ixj > i) {
          bool up = (i & kk) == 0;
          unsigned long long a = k[i], c = k[ixj];
          if ((a > c) == up) { k[i] = c; k[ixj] = a; }
        }
      }
      __syncthreads();
    }
  }
  int local = 0;
  for (int j = tid; j < N; j += 1024) {
    unsigned long long key = k[j];
    int idx = (int)(key & 0xFFFFFFFFull);
    int src = b * N + idx, dst = b * N + j;
    boxes_s[dst] = boxes_u[src];
    cls_s[dst]   = cls_u[src];
    obj_s[dst]   = obj_u[src];
    conf_s[dst]  = conf_u[src];
    if ((unsigned)(key >> 32) != 0xFF800000u) local++;
  }
  atomicAdd(&cnt, local);
  __syncthreads();
  if (tid == 0) Vout[b] = cnt;
}

__device__ __forceinline__ unsigned onehot4(int c, int base) {
  unsigned d = (unsigned)(c - base);
  return (d < 4u) ? (1u << (d * 8)) : 0u;
}

__global__ void supmat_kernel(const float4* __restrict__ boxes_s,
                              const int* __restrict__ cls_s,
                              const int* __restrict__ Vp,
                              unsigned* __restrict__ sup) {
  int cw = blockIdx.x;
  int rt = blockIdx.y;
  int b  = blockIdx.z;
  int V = Vp[b];
  int r0 = rt * 16, c0 = cw * 32;
  if (r0 >= V || c0 >= V) return;
  int lane = threadIdx.x;
  int h = lane >> 4;
  int m = lane & 15;

  __shared__ float4 rbox[16];
  __shared__ float4 cbox[32];
  if (lane < 16) rbox[lane] = boxes_s[b * N + r0 + lane];
  cbox[lane] = boxes_s[b * N + c0 + lane];

  int clsRow = cls_s[b * N + r0 + m];
  int clsC0  = cls_s[b * N + c0 + m];
  int clsC1  = cls_s[b * N + c0 + 16 + m];

  v8i a0, a1, b00, b01, b10, b11;
#pragma unroll
  for (int v = 0; v < 8; ++v) {
    int baseA = ((v & 3) << 2) | (h << 4) | ((v >> 2) << 5);
    a0[v]  = (int)onehot4(clsRow,      baseA);
    a1[v]  = (int)onehot4(clsRow - 64, baseA);
    int baseB = ((v & 3) << 2) | (h << 4) | ((v >> 2) << 5);
    b00[v] = (int)onehot4(clsC0,      baseB);
    b01[v] = (int)onehot4(clsC0 - 64, baseB);
    b10[v] = (int)onehot4(clsC1,      baseB);
    b11[v] = (int)onehot4(clsC1 - 64, baseB);
  }
  v8i acc0 = {0,0,0,0,0,0,0,0}, acc1 = {0,0,0,0,0,0,0,0};
  acc0 = __builtin_amdgcn_wmma_i32_16x16x64_iu8(false, a0, false, b00, acc0, false, false);
  acc0 = __builtin_amdgcn_wmma_i32_16x16x64_iu8(false, a1, false, b01, acc0, false, false);
  acc1 = __builtin_amdgcn_wmma_i32_16x16x64_iu8(false, a0, false, b10, acc1, false, false);
  acc1 = __builtin_amdgcn_wmma_i32_16x16x64_iu8(false, a1, false, b11, acc1, false, false);

  __syncthreads();

  float4 cb0 = cbox[m];
  float4 cb1 = cbox[16 + m];
  float areaC0 = (cb0.z - cb0.x) * (cb0.w - cb0.y);
  float areaC1 = (cb1.z - cb1.x) * (cb1.w - cb1.y);

  size_t rowBase = ((size_t)b * N + (size_t)r0) * W + cw;
#pragma unroll
  for (int r = 0; r < 8; ++r) {
    float4 rb = rbox[r + 8 * h];
    float areaR = (rb.z - rb.x) * (rb.w - rb.y);

    float ix1 = fmaxf(rb.x, cb0.x), iy1 = fmaxf(rb.y, cb0.y);
    float ix2 = fminf(rb.z, cb0.z), iy2 = fminf(rb.w, cb0.w);
    float inter = fmaxf(ix2 - ix1, 0.0f) * fmaxf(iy2 - iy1, 0.0f);
    float iou = inter / (areaR + areaC0 - inter + 1e-16f);
    unsigned bal0 = __builtin_amdgcn_ballot_w32((acc0[r] != 0) && (iou > NMS_TH));

    ix1 = fmaxf(rb.x, cb1.x); iy1 = fmaxf(rb.y, cb1.y);
    ix2 = fminf(rb.z, cb1.z); iy2 = fminf(rb.w, cb1.w);
    inter = fmaxf(ix2 - ix1, 0.0f) * fmaxf(iy2 - iy1, 0.0f);
    iou = inter / (areaR + areaC1 - inter + 1e-16f);
    unsigned bal1 = __builtin_amdgcn_ballot_w32((acc1[r] != 0) && (iou > NMS_TH));

    if (lane == 0) {
      sup[rowBase + (size_t)r * W]       = (bal0 & 0xFFFFu) | (bal1 << 16);
      sup[rowBase + (size_t)(r + 8) * W] = (bal0 >> 16) | (bal1 & 0xFFFF0000u);
    }
  }
}

__global__ void scan_kernel(const unsigned* __restrict__ sup,
                            const int* __restrict__ Vp,
                            unsigned* __restrict__ keepOut) {
  int b = blockIdx.x;
  int lane = threadIdx.x;
  int V = Vp[b];
  const unsigned* base = sup + (size_t)b * N * W;
  uint4 supp = make_uint4(0, 0, 0, 0);
  uint4 row  = make_uint4(0, 0, 0, 0);
  if (V > 0) row = *(const uint4*)(base + (size_t)lane * 4);
  for (int i = 0; i < V; ++i) {
    if (i + 2 < V)
      __builtin_prefetch(base + (size_t)(i + 2) * W + lane * 4, 0, 0);
    uint4 nrow = make_uint4(0, 0, 0, 0);
    if (i + 1 < V) nrow = *(const uint4*)(base + (size_t)(i + 1) * W + lane * 4);
    int owner = i >> 7;
    int sel = (i >> 5) & 3;
    unsigned myw = sel == 0 ? supp.x : sel == 1 ? supp.y : sel == 2 ? supp.z : supp.w;
    unsigned wv = __shfl(myw, owner, 32);
    unsigned keep = ((wv >> (i & 31)) & 1u) ^ 1u;
    if (keep) {
      supp.x |= row.x; supp.y |= row.y; supp.z |= row.z; supp.w |= row.w;
    }
    if (lane == 0) keepOut[b * N + i] = keep;
    row = nrow;
  }
}

__global__ void out_kernel(const float4* __restrict__ boxes_s,
                           const int* __restrict__ cls_s,
                           const float* __restrict__ obj_s,
                           const float* __restrict__ conf_s,
                           const int* __restrict__ Vp,
                           const unsigned* __restrict__ keepArr,
                           float* __restrict__ out) {
  int g = blockIdx.x * blockDim.x + threadIdx.x;
  if (g >= BATCH * N) return;
  int b = g >> 12;
  int j = g & (N - 1);
  bool kept = (j < Vp[b]) && (keepArr[g] != 0);
  float4 bx = make_float4(0, 0, 0, 0);
  float ob = 0, cf = 0, cl = 0, bi = 0;
  if (kept) {
    bx = boxes_s[g]; ob = obj_s[g]; cf = conf_s[g];
    cl = (float)cls_s[g]; bi = (float)b;
  }
  float4* o = (float4*)(out + (size_t)g * 8);
  o[0] = make_float4(bi, bx.x, bx.y, bx.z);
  o[1] = make_float4(bx.w, ob, cf, cl);
}

extern "C" void kernel_launch(void* const* d_in, const int* in_sizes, int n_in,
                              void* d_out, int out_size, void* d_ws, size_t ws_size,
                              hipStream_t stream) {
  (void)in_sizes; (void)n_in; (void)out_size; (void)ws_size;
  const float* x = (const float*)d_in[0];
  float* out = (float*)d_out;

  char* w = (char*)d_ws;
  auto alloc = [&](size_t bytes) -> void* {
    void* p = (void*)w;
    w += (bytes + 255) & ~(size_t)255;
    return p;
  };
  const size_t BN = (size_t)BATCH * N;
  float4* boxes_u = (float4*)alloc(BN * sizeof(float4));
  float4* boxes_s = (float4*)alloc(BN * sizeof(float4));
  int*    cls_u   = (int*)alloc(BN * 4);
  int*    cls_s   = (int*)alloc(BN * 4);
  float*  obj_u   = (float*)alloc(BN * 4);
  float*  obj_s   = (float*)alloc(BN * 4);
  float*  conf_u  = (float*)alloc(BN * 4);
  float*  conf_s  = (float*)alloc(BN * 4);
  unsigned long long* keys = (unsigned long long*)alloc(BN * 8);
  int*      Vp   = (int*)alloc(256);
  unsigned* keep = (unsigned*)alloc(BN * 4);
  unsigned* sup  = (unsigned*)alloc(BN * (size_t)W * 4);

  prep_kernel<<<dim3((unsigned)(BN / 4)), 128, 0, stream>>>(
      x, boxes_u, cls_u, obj_u, conf_u, keys);
  sort_kernel<<<dim3(BATCH), 1024, 0, stream>>>(
      keys, boxes_u, cls_u, obj_u, conf_u, boxes_s, cls_s, obj_s, conf_s, Vp);
  supmat_kernel<<<dim3(W, N / 16, BATCH), 32, 0, stream>>>(boxes_s, cls_s, Vp, sup);
  scan_kernel<<<dim3(BATCH), 32, 0, stream>>>(sup, Vp, keep);
  out_kernel<<<dim3((unsigned)((BN + 255) / 256)), 256, 0, stream>>>(
      boxes_s, cls_s, obj_s, conf_s, Vp, keep, out);
}
